// PnPAttentionProcessor_54262616818232
// MI455X (gfx1250) — compile-verified, matrix operand order checked by reading, not run on hardware
//
#include <hip/hip_runtime.h>
#include <hip/hip_bf16.h>

// ---------------------------------------------------------------------------
// PnP attention, MI455X (gfx1250, wave32, WMMA bf16).
//   B=8, S=1024, C=1280, H=20, D=64.  Only batches {0,4} are unique after
//   injection; compute those, broadcast 4x at the end.
// Pipeline: cvt(x,W->bf16)  ->  fused QKV GEMM (WMMA)  ->  flash attention
//           (WMMA QK^T + online softmax + WMMA PV)     ->  O-proj GEMM + bias
// ---------------------------------------------------------------------------

typedef __attribute__((ext_vector_type(16))) __bf16 v16bf;
typedef __attribute__((ext_vector_type(8)))  float  v8f;
typedef unsigned short u16;

#define S_  1024
#define C_  1280
#define H_  20
#define D_  64
#define NB_ 2          // unique batches (0 and 4)

// ---- fp32 -> bf16 (round to nearest even) ---------------------------------
__device__ __forceinline__ u16 f2bf(float f) {
  unsigned u = __builtin_bit_cast(unsigned, f);
  u = (u + 0x7FFFu + ((u >> 16) & 1u)) >> 16;
  return (u16)u;
}

// ---- load a 16x32 bf16 fragment: two contiguous 16-byte runs --------------
__device__ __forceinline__ v16bf load_frag(const u16* p0, const u16* p1) {
  union { v16bf v; uint4 u[2]; } f;
  f.u[0] = *reinterpret_cast<const uint4*>(p0);   // elements 0..7  (k = kb..kb+7)
  f.u[1] = *reinterpret_cast<const uint4*>(p1);   // elements 8..15 (k = kb+16..kb+23)
  return f.v;
}

__device__ __forceinline__ v8f wmma_bf16(v16bf a, v16bf b, v8f c) {
  return __builtin_amdgcn_wmma_f32_16x16x32_bf16(
      /*neg_a=*/false, a, /*neg_b=*/false, b,
      /*c_mod=*/(short)0, c, /*reuse_a=*/false, /*reuse_b=*/false);
}

// ---- 16-lane half-wave reductions (C/D rows live per half) ----------------
__device__ __forceinline__ float hmax16(float v) {
#pragma unroll
  for (int off = 1; off < 16; off <<= 1) v = fmaxf(v, __shfl_xor(v, off, 16));
  return v;
}
__device__ __forceinline__ float hsum16(float v) {
#pragma unroll
  for (int off = 1; off < 16; off <<= 1) v += __shfl_xor(v, off, 16);
  return v;
}

// ---------------------------------------------------------------------------
// conversions
// ---------------------------------------------------------------------------
__global__ void cvt_x_kernel(const float* __restrict__ hs, u16* __restrict__ x2) {
  int i = blockIdx.x * blockDim.x + threadIdx.x;          // NB_*S_*C_ exact
  int half = i / (S_ * C_);
  int rem  = i - half * (S_ * C_);
  x2[i] = f2bf(hs[(size_t)(half * 4) * (S_ * C_) + rem]); // batches 0 and 4
}

__global__ void cvt_wT_kernel(const float* __restrict__ w, u16* __restrict__ wT) {
  int i = blockIdx.x * blockDim.x + threadIdx.x;          // C_*C_ exact
  int n = i / C_, k = i - n * C_;
  wT[i] = f2bf(w[(size_t)k * C_ + n]);                    // wT[n][k] = w[k][n]
}

// ---------------------------------------------------------------------------
// 64x64-per-wave GEMM inner loop: A [.. x K] row-major bf16 (lane=M rows),
// B^T [.. x K] row-major bf16 (lane=N cols). 16 WMMAs per K=32 step.
// ---------------------------------------------------------------------------
__device__ __forceinline__ void gemm64x64(const u16* __restrict__ A,
                                          const u16* __restrict__ B,
                                          int K, int ln, int kbo, v8f acc[16]) {
  for (int kk = 0; kk < K; kk += 32) {
    v16bf a[4], b[4];
#pragma unroll
    for (int i = 0; i < 4; ++i) {
      const u16* pa = A + (size_t)(i * 16 + ln) * K + kk + kbo;
      a[i] = load_frag(pa, pa + 16);
    }
#pragma unroll
    for (int j = 0; j < 4; ++j) {
      const u16* pb = B + (size_t)(j * 16 + ln) * K + kk + kbo;
      b[j] = load_frag(pb, pb + 16);
    }
    if (kk + 32 < K) {  // -> global_prefetch_b8
      __builtin_prefetch(A + (size_t)ln * K + kk + 32, 0, 1);
      __builtin_prefetch(B + (size_t)ln * K + kk + 32, 0, 1);
    }
#pragma unroll
    for (int i = 0; i < 4; ++i)
#pragma unroll
      for (int j = 0; j < 4; ++j)
        acc[i * 4 + j] = wmma_bf16(a[i], b[j], acc[i * 4 + j]);
  }
}

// ---------------------------------------------------------------------------
// fused QKV: X2[2048 x 1280] @ {Wq,Wk,Wv}^T.  1920 wave-tiles (w,mt,nt).
// Q scaled by 1/sqrt(D).  Q,K -> [b][h][s][d]; V -> [b][h][d][s] (transposed).
// ---------------------------------------------------------------------------
__global__ void __launch_bounds__(256) qkv_gemm_kernel(
    const u16* __restrict__ x2,
    const u16* __restrict__ wqT, const u16* __restrict__ wkT,
    const u16* __restrict__ wvT,
    u16* __restrict__ qb, u16* __restrict__ kb, u16* __restrict__ vt) {
  const int lane = threadIdx.x & 31, wave = threadIdx.x >> 5;
  const int ln = lane & 15, hf = lane >> 4, kbo = hf * 8;
  int t = blockIdx.x * 8 + wave;                 // 0..1919
  const int w = t / 640; t -= w * 640;
  const int mt = t / 20, nt = t - mt * 20;       // 32 x 20 tiles of 64x64
  const u16* wT = (w == 0) ? wqT : (w == 1) ? wkT : wvT;

  v8f acc[16];
#pragma unroll
  for (int i = 0; i < 16; ++i) acc[i] = (v8f){};

  gemm64x64(x2 + (size_t)(mt * 64) * C_, wT + (size_t)(nt * 64) * C_,
            C_, ln, kbo, acc);

  const float qscale = 0.125f;                   // 1/sqrt(64)
#pragma unroll
  for (int i = 0; i < 4; ++i)
#pragma unroll
    for (int j = 0; j < 4; ++j)
#pragma unroll
      for (int r = 0; r < 8; ++r) {
        int m = mt * 64 + i * 16 + hf * 8 + r;   // row (b*S+s)
        int n = nt * 64 + j * 16 + ln;           // col (h*D+d)
        int bi = m >> 10, s = m & 1023;
        int h = n >> 6, d = n & 63;
        float v = acc[i * 4 + j][r];
        if (w == 0) v *= qscale;
        u16 bv = f2bf(v);
        if (w == 0)      qb[(((size_t)bi * H_ + h) * S_ + s) * D_ + d] = bv;
        else if (w == 1) kb[(((size_t)bi * H_ + h) * S_ + s) * D_ + d] = bv;
        else             vt[(((size_t)bi * H_ + h) * D_ + d) * S_ + s] = bv;
      }
}

// ---------------------------------------------------------------------------
// flash attention: one wave = one 16-query tile of one (b,h).
// grid = 40 (b,h) * 8 blocks, 8 waves/block -> 64 q-tiles per (b,h).
// ---------------------------------------------------------------------------
__global__ void __launch_bounds__(256) attn_kernel(
    const u16* __restrict__ qb, const u16* __restrict__ kb,
    const u16* __restrict__ vt, u16* __restrict__ ob) {
  __shared__ __align__(16) u16 plds[8 * 16 * 32];  // 1KB P-staging per wave
  const int lane = threadIdx.x & 31, wave = threadIdx.x >> 5;
  const int ln = lane & 15, hf = lane >> 4, kbo = hf * 8;
  const int bh = blockIdx.x >> 3;                  // 0..39
  const int qtile = (blockIdx.x & 7) * 8 + wave;   // 0..63
  const int bi = bh / H_, h = bh - bi * H_;
  const int q0 = qtile * 16;
  u16* lb = plds + wave * 512;

  // Q fragments (A-layout: lane=M row, interleaved K runs), held for the loop
  const u16* qrow = qb + ((size_t)bh * S_ + (q0 + ln)) * D_;
  const v16bf qa0 = load_frag(qrow + kbo,      qrow + kbo + 16);       // d 0..31
  const v16bf qa1 = load_frag(qrow + 32 + kbo, qrow + 32 + kbo + 16);  // d 32..63

  float mrow[8], lrow[8];
  v8f acc[4];
#pragma unroll
  for (int r = 0; r < 8; ++r) { mrow[r] = -1e30f; lrow[r] = 0.0f; }
#pragma unroll
  for (int g = 0; g < 4; ++g) acc[g] = (v8f){};

  for (int j = 0; j < S_ / 32; ++j) {
    // ---- S = Q K^T (Q pre-scaled): two 16x16 C/D tiles over 32 keys ----
    v8f s[2];
#pragma unroll
    for (int tt = 0; tt < 2; ++tt) {
      const u16* krow = kb + ((size_t)bh * S_ + (j * 32 + tt * 16 + ln)) * D_;
      v16bf kf0 = load_frag(krow + kbo,      krow + kbo + 16);
      v16bf kf1 = load_frag(krow + 32 + kbo, krow + 32 + kbo + 16);
      v8f z = (v8f){};
      s[tt] = wmma_bf16(qa0, kf0, z);
      s[tt] = wmma_bf16(qa1, kf1, s[tt]);
    }
    // ---- online softmax (rows live per half-wave; stats per C/D slot) ----
    float corr[8];
#pragma unroll
    for (int r = 0; r < 8; ++r) {
      float v = hmax16(fmaxf(s[0][r], s[1][r]));
      float mn = fmaxf(mrow[r], v);
      corr[r] = __expf(mrow[r] - mn);
      mrow[r] = mn;
      float p0 = __expf(s[0][r] - mn);
      float p1 = __expf(s[1][r] - mn);
      lrow[r] = lrow[r] * corr[r] + hsum16(p0 + p1);
      lb[(hf * 8 + r) * 32 + ln]      = f2bf(p0);   // P tile -> LDS, row-major
      lb[(hf * 8 + r) * 32 + 16 + ln] = f2bf(p1);
    }
#pragma unroll
    for (int g = 0; g < 4; ++g)
#pragma unroll
      for (int r = 0; r < 8; ++r) acc[g][r] *= corr[r];

    // wave-local: drain DS stores before re-reading P as an A-fragment
    asm volatile("s_wait_dscnt 0" ::: "memory");
    const u16* prow = lb + ln * 32;
    v16bf pf = load_frag(prow + kbo, prow + kbo + 16);   // P 16x32 A-frag

    // ---- O += P V  (V transposed: contiguous key runs per d-column) ----
#pragma unroll
    for (int g = 0; g < 4; ++g) {
      const u16* vrow = vt + ((size_t)bh * D_ + g * 16 + ln) * S_ + j * 32;
      v16bf vf = load_frag(vrow + kbo, vrow + kbo + 16);
      acc[g] = wmma_bf16(pf, vf, acc[g]);
    }
  }

  // ---- normalize, emit O as bf16 [b][s][h*D+d] for the projection GEMM ----
#pragma unroll
  for (int r = 0; r < 8; ++r) {
    float inv = 1.0f / lrow[r];
    int srow = q0 + hf * 8 + r;
#pragma unroll
    for (int g = 0; g < 4; ++g) {
      int c = h * 64 + g * 16 + ln;
      ob[((size_t)bi * S_ + srow) * C_ + c] = f2bf(acc[g][r] * inv);
    }
  }
}

// ---------------------------------------------------------------------------
// output projection: O[2048 x 1280] @ Wo^T + bo, broadcast each unique batch
// to its 4 replicas in the fp32 output.
// ---------------------------------------------------------------------------
__global__ void __launch_bounds__(256) proj_gemm_kernel(
    const u16* __restrict__ ob, const u16* __restrict__ woT,
    const float* __restrict__ bo, float* __restrict__ out) {
  const int lane = threadIdx.x & 31, wave = threadIdx.x >> 5;
  const int ln = lane & 15, hf = lane >> 4, kbo = hf * 8;
  int t = blockIdx.x * 8 + wave;                 // 0..639
  const int mt = t / 20, nt = t - mt * 20;

  v8f acc[16];
#pragma unroll
  for (int i = 0; i < 16; ++i) acc[i] = (v8f){};

  gemm64x64(ob + (size_t)(mt * 64) * C_, woT + (size_t)(nt * 64) * C_,
            C_, ln, kbo, acc);

#pragma unroll
  for (int j = 0; j < 4; ++j) {
    int n = nt * 64 + j * 16 + ln;
    float bias = bo[n];
#pragma unroll
    for (int i = 0; i < 4; ++i)
#pragma unroll
      for (int r = 0; r < 8; ++r) {
        int m = mt * 64 + i * 16 + hf * 8 + r;
        int bi = m >> 10, s = m & 1023;
        float y = acc[i * 4 + j][r] + bias;
#pragma unroll
        for (int rep = 0; rep < 4; ++rep)
          out[(((size_t)(bi * 4 + rep)) * S_ + s) * C_ + n] = y;
      }
  }
}

// ---------------------------------------------------------------------------
// launch
// ---------------------------------------------------------------------------
extern "C" void kernel_launch(void* const* d_in, const int* in_sizes, int n_in,
                              void* d_out, int out_size, void* d_ws, size_t ws_size,
                              hipStream_t stream) {
  const float* hs = (const float*)d_in[0];
  const float* Wq = (const float*)d_in[1];
  const float* Wk = (const float*)d_in[2];
  const float* Wv = (const float*)d_in[3];
  const float* Wo = (const float*)d_in[4];
  const float* bo = (const float*)d_in[5];
  float* out = (float*)d_out;

  // workspace layout (bf16 elements), ~40 MB total
  u16* ws  = (u16*)d_ws;
  u16* x2  = ws;                                  // [2][1024][1280]
  u16* wqT = x2  + (size_t)NB_ * S_ * C_;         // [1280][1280] each
  u16* wkT = wqT + (size_t)C_ * C_;
  u16* wvT = wkT + (size_t)C_ * C_;
  u16* woT = wvT + (size_t)C_ * C_;
  u16* qb  = woT + (size_t)C_ * C_;               // [2][20][1024][64]
  u16* kb  = qb  + (size_t)NB_ * H_ * S_ * D_;
  u16* vt  = kb  + (size_t)NB_ * H_ * S_ * D_;    // [2][20][64][1024]
  u16* ob  = vt  + (size_t)NB_ * H_ * S_ * D_;    // [2][1024][1280]

  cvt_x_kernel<<<(NB_ * S_ * C_) / 256, 256, 0, stream>>>(hs, x2);
  cvt_wT_kernel<<<(C_ * C_) / 256, 256, 0, stream>>>(Wq, wqT);
  cvt_wT_kernel<<<(C_ * C_) / 256, 256, 0, stream>>>(Wk, wkT);
  cvt_wT_kernel<<<(C_ * C_) / 256, 256, 0, stream>>>(Wv, wvT);
  cvt_wT_kernel<<<(C_ * C_) / 256, 256, 0, stream>>>(Wo, woT);

  // 3 mats * 32*20 tiles = 1920 wave-tiles, 8 waves/block
  qkv_gemm_kernel<<<240, 256, 0, stream>>>(x2, wqT, wkT, wvT, qb, kb, vt);

  // 40 (b,h) * 8 blocks, 8 waves/block -> 64 q-tiles each
  attn_kernel<<<320, 256, 0, stream>>>(qb, kb, vt, ob);

  // 32*20 = 640 wave-tiles
  proj_gemm_kernel<<<80, 256, 0, stream>>>(ob, woT, bo, out);
}
